// S4Block_49898930045305
// MI455X (gfx1250) — matrix operand order repaired and checked by reading, not compiled, not run on hardware
//
#include <hip/hip_runtime.h>
#include <hip/hip_bf16.h>

// ---------------------------------------------------------------------------
// S4-style block for MI455X (gfx1250, wave32, WMMA).
//   Stage P:  pre-pack B, C into bf16 WMMA-operand layout (512 KB, once/call)
//   Stage A:  xB = x @ B^T  (bf16 WMMA, f32 acc, ping-pong pipelined operands)
//   Stage B:  state_t = tanh(state_{t-1} @ A^T + xB_t)  (single-WGP WMMA scan)
//   Stage C:  y = states @ C^T + (D+1)*x ; LayerNorm(y) (bf16 WMMA + fused LN)
// ---------------------------------------------------------------------------

#define D_MODEL 1024
#define D_STATE 128
#define BATCH   16
#define SEQ     2048
#define LN_EPS  1e-5f

typedef __attribute__((ext_vector_type(16))) __bf16         v16bf;
typedef __attribute__((ext_vector_type(2)))  __bf16         v2bf;
typedef __attribute__((ext_vector_type(8)))  float          v8f;
typedef __attribute__((ext_vector_type(16))) unsigned short us16;
typedef __attribute__((ext_vector_type(8)))  unsigned short us8;
typedef __attribute__((ext_vector_type(8)))  unsigned int   u32x8;
typedef __attribute__((ext_vector_type(4)))  float          f32x4;

// f32 -> bf16 RNE (cold paths only)
__device__ __forceinline__ unsigned short f2bf(float f) {
  unsigned u = __builtin_bit_cast(unsigned, f);
  unsigned r = u + 0x7FFFu + ((u >> 16) & 1u);
  return (unsigned short)(r >> 16);
}

// packed pair convert, element 0 -> low half. Hot path: 2x v_add + 1x v_perm.
__device__ __forceinline__ unsigned pk2bf(float lo, float hi) {
#if __has_builtin(__builtin_amdgcn_cvt_pk_bf16_f32)
  v2bf p = __builtin_amdgcn_cvt_pk_bf16_f32(lo, hi);
  return __builtin_bit_cast(unsigned, p);
#elif __has_builtin(__builtin_amdgcn_perm)
  unsigned ul = __builtin_bit_cast(unsigned, lo) + 0x8000u;  // round-half-up
  unsigned uh = __builtin_bit_cast(unsigned, hi) + 0x8000u;
  return __builtin_amdgcn_perm(uh, ul, 0x07060302u);         // {uh[31:16],ul[31:16]}
#else
  return (unsigned)f2bf(lo) | ((unsigned)f2bf(hi) << 16);
#endif
}

__device__ __forceinline__ v8f wmma_bf16(us16 a, us16 b, v8f c) {
  return __builtin_amdgcn_wmma_f32_16x16x32_bf16(
      false, __builtin_bit_cast(v16bf, a),
      false, __builtin_bit_cast(v16bf, b),
      (short)0, c, false, false);
}

__device__ __forceinline__ float fast_tanh(float x) {
#if __has_builtin(__builtin_amdgcn_tanhf)
  return __builtin_amdgcn_tanhf(x);
#else
  float e = __builtin_amdgcn_exp2f(x * 2.8853900817779268f);
  return 1.0f - 2.0f * __builtin_amdgcn_rcpf(1.0f + e);
#endif
}

// Operand layout conventions (wave32, CDNA5 ISA):
//  A(16x32 bf16): lane m=l%16 is row; lanes<16 hold K 0..15,
//                 lanes>=16 hold K 16..31 of each 32-K chunk (same as B).
//  B(32x16 bf16): lane n=l%16 is col; lanes<16 hold K 0..15, lanes>=16 K 16..31.
//  C/D(16x16 f32): VGPR r, lane l -> row r + 8*(l/16), col l%16.
//
// xB workspace uses accumulator-native tiles: xBs[((t*8+tile)*32+lane)*8 + r],
// so stage A stores one v8f per tile and the scan loads it back verbatim.

#define BOP_N (32 * 8 * 32 * 16)
#define COP_N (64 * 4 * 32 * 16)

// ---------------------------------------------------------------------------
// Stage P: pack B [128][1024] and C [1024][128] into bf16 operand tiles.
//  Bop tile (kc,w):  idx = ((kc*8+w)*32 + lane)*16 + e ; n=16w+l%16, k=32kc+e+16*(l/16)
//  Cop tile (nt,kc): idx = ((nt*4+kc)*32 + lane)*16 + e ; d=16nt+l%16, k=32kc+e+16*(l/16)
// ---------------------------------------------------------------------------
__global__ __launch_bounds__(256) void kP_pack(
    const float* __restrict__ Bm, const float* __restrict__ Cm,
    unsigned short* __restrict__ Bop, unsigned short* __restrict__ Cop) {
  const int stride = gridDim.x * 256;
  for (int i = blockIdx.x * 256 + threadIdx.x; i < BOP_N; i += stride) {
    int e = i & 15, lane = (i >> 4) & 31, tile = i >> 9;   // tile = kc*8+w
    int kc = tile >> 3, w = tile & 7;
    int n = w * 16 + (lane & 15);
    int k = kc * 32 + e + 16 * (lane >> 4);
    Bop[i] = f2bf(Bm[(size_t)n * D_MODEL + k]);
  }
  for (int i = blockIdx.x * 256 + threadIdx.x; i < COP_N; i += stride) {
    int e = i & 15, lane = (i >> 4) & 31, tile = i >> 9;   // tile = nt*4+kc
    int kc = tile & 3, nt = tile >> 2;
    int d = nt * 16 + (lane & 15);
    int k = kc * 32 + e + 16 * (lane >> 4);
    Cop[i] = f2bf(Cm[(size_t)d * D_STATE + k]);
  }
}

// ---------------------------------------------------------------------------
// Stage A: grid = SEQ/8 blocks, 256 threads; each WAVE owns one timestep t
// (M = 16 batches) and all N = 128 state columns (8 accumulators).
// Operands ping-pong double-buffered: chunk kc+1 loads issue before chunk
// kc's 8 WMMAs, so L2 latency overlaps matrix work.
// ---------------------------------------------------------------------------
__global__ __launch_bounds__(256) void kA_xproj(
    const float* __restrict__ x,            // [BATCH][SEQ][D_MODEL]
    const unsigned short* __restrict__ Bop, // packed
    float* __restrict__ xBs) {              // tiles, see above
  const int lane = threadIdx.x & 31;
  const int wave = threadIdx.x >> 5;
  const int t    = blockIdx.x * 8 + wave;
  const int m    = lane & 15;
  const int half = lane >> 4;

  v8f acc[8];
#pragma unroll
  for (int w2 = 0; w2 < 8; ++w2)
    acc[w2] = (v8f){0.f, 0.f, 0.f, 0.f, 0.f, 0.f, 0.f, 0.f};

  const float* xrow = x + ((size_t)m * SEQ + t) * D_MODEL;  // batch = m

  auto ldA = [&](int kc) -> us16 {
    const int ka = kc * 32 + 16 * half;
    f32x4 a0 = *(const f32x4*)(xrow + ka);
    f32x4 a1 = *(const f32x4*)(xrow + ka + 4);
    f32x4 a2 = *(const f32x4*)(xrow + ka + 8);
    f32x4 a3 = *(const f32x4*)(xrow + ka + 12);
    u32x8 aw;
    aw[0] = pk2bf(a0[0], a0[1]); aw[1] = pk2bf(a0[2], a0[3]);
    aw[2] = pk2bf(a1[0], a1[1]); aw[3] = pk2bf(a1[2], a1[3]);
    aw[4] = pk2bf(a2[0], a2[1]); aw[5] = pk2bf(a2[2], a2[3]);
    aw[6] = pk2bf(a3[0], a3[1]); aw[7] = pk2bf(a3[2], a3[3]);
    return __builtin_bit_cast(us16, aw);
  };
  auto ldB = [&](int kc, us16* b) {
#pragma unroll
    for (int w2 = 0; w2 < 8; ++w2)
      b[w2] = *(const us16*)(Bop + ((size_t)(kc * 8 + w2) * 32 + lane) * 16);
  };

  us16 avA, avB, bvA[8], bvB[8];
  avA = ldA(0); ldB(0, bvA);
  for (int kc = 0; kc < 32; kc += 2) {
    avB = ldA(kc + 1); ldB(kc + 1, bvB);          // prefetch odd chunk
#pragma unroll
    for (int w2 = 0; w2 < 8; ++w2)
      acc[w2] = wmma_bf16(avA, bvA[w2], acc[w2]);
    if (kc + 2 < 32) { avA = ldA(kc + 2); ldB(kc + 2, bvA); }  // prefetch even
#pragma unroll
    for (int w2 = 0; w2 < 8; ++w2)
      acc[w2] = wmma_bf16(avB, bvB[w2], acc[w2]);
  }

#pragma unroll
  for (int w2 = 0; w2 < 8; ++w2)
    *(v8f*)(xBs + ((size_t)(t * 8 + w2) * 32 + lane) * 8) = acc[w2];
}

// ---------------------------------------------------------------------------
// Stage B: sequential scan. ONE workgroup, 8 waves; wave w owns state cols
// [16w,16w+16). bf16 state in double-buffered LDS; ONE barrier per step;
// xB_t register-prefetched 8 steps deep (2x b128 each); 2x2 WMMA chains.
// ---------------------------------------------------------------------------
#define SROW 136

__global__ __launch_bounds__(256) void kB_scan(
    const float* __restrict__ Am,           // [D_STATE][D_STATE]
    const float* __restrict__ xBs,          // tiles
    unsigned short* __restrict__ statesbf) {// [SEQ][BATCH][D_STATE] bf16
  const int lane = threadIdx.x & 31;
  const int wave = threadIdx.x >> 5;
  const int m    = lane & 15;
  const int half = lane >> 4;
  const int ncol = wave * 16 + m;

  __shared__ __align__(16) unsigned short st[2][BATCH][SROW];
  for (int i = threadIdx.x; i < BATCH * SROW; i += 256)
    (&st[0][0][0])[i] = 0;                  // initial state = 0 (buffer 0)

  // B operand = A^T in registers: state_new[m,n] = sum_k state[m,k]*A[n,k]
  us16 bop[4];
#pragma unroll
  for (int kc = 0; kc < 4; ++kc) {
    const float* arow = Am + (size_t)ncol * D_STATE + kc * 32 + 16 * half;
    u32x8 w;
#pragma unroll
    for (int p = 0; p < 8; ++p) w[p] = pk2bf(arow[2 * p], arow[2 * p + 1]);
    bop[kc] = __builtin_bit_cast(us16, w);
  }
  __syncthreads();

  auto ldxb = [&](v8f* c, int t) {          // wave's own tile: 2x b128
    *c = *(const v8f*)(xBs + ((size_t)(t * 8 + wave) * 32 + lane) * 8);
  };

  auto step = [&](int t, v8f c) {
    const int rb = t & 1, wb = rb ^ 1;
    v8f accA = c, accB = (v8f){0.f, 0.f, 0.f, 0.f, 0.f, 0.f, 0.f, 0.f};
    us16 av[4];
#pragma unroll
    for (int kc = 0; kc < 4; ++kc) {
      const int ka = kc * 32 + 16 * half;
      us8 lo = *(const us8*)&st[rb][m][ka];         // ds_load_b128
      us8 hi = *(const us8*)&st[rb][m][ka + 8];     // ds_load_b128
#pragma unroll
      for (int e = 0; e < 8; ++e) { av[kc][e] = lo[e]; av[kc][e + 8] = hi[e]; }
    }
    // two independent 2-deep WMMA chains, then combine
    accA = wmma_bf16(av[0], bop[0], accA);
    accB = wmma_bf16(av[1], bop[1], accB);
    accA = wmma_bf16(av[2], bop[2], accA);
    accB = wmma_bf16(av[3], bop[3], accB);
    unsigned short* orow = statesbf + (size_t)t * (BATCH * D_STATE);
#pragma unroll
    for (int r = 0; r < 8; ++r) {
      float s = fast_tanh(accA[r] + accB[r]);
      const int row = r + 8 * half;
      unsigned u = __builtin_bit_cast(unsigned, s) + 0x8000u;  // round-half-up
      unsigned short sb = (unsigned short)(u >> 16);           // d16_hi-friendly
      st[wb][row][ncol] = sb;                       // next-step state
      orow[row * D_STATE + ncol] = sb;              // stage-C input
    }
    __syncthreads();                                // write(t) -> read(t+1)
  };

  v8f cb[8];
#pragma unroll
  for (int s = 0; s < 8; ++s) ldxb(&cb[s], s);
  for (int t = 0; t < SEQ; t += 8) {
#pragma unroll
    for (int s = 0; s < 8; ++s) {
      step(t + s, cb[s]);
      if (t + s + 8 < SEQ) ldxb(&cb[s], t + s + 8);
    }
  }
}

// ---------------------------------------------------------------------------
// Stage C: out = states @ C^T + (D+1)*x, fused LayerNorm. grid = SEQ blocks,
// 256 threads; wave w owns d-columns [128w,128w+128) as 8 N-subtiles.
// ---------------------------------------------------------------------------
__global__ __launch_bounds__(256) void kC_out_ln(
    const unsigned short* __restrict__ statesbf, // [SEQ][BATCH][D_STATE] bf16
    const float* __restrict__ x,       // [BATCH][SEQ][D_MODEL]
    const unsigned short* __restrict__ Cop,      // packed
    const float* __restrict__ Dv,      // [D_MODEL]
    const float* __restrict__ gamma,   // [D_MODEL]
    const float* __restrict__ beta,    // [D_MODEL]
    float* __restrict__ out) {         // [BATCH][SEQ][D_MODEL]
  const int t    = blockIdx.x;
  const int lane = threadIdx.x & 31;
  const int wave = threadIdx.x >> 5;
  const int m    = lane & 15;
  const int half = lane >> 4;

  __shared__ float s_sum[BATCH];
  __shared__ float s_ssq[BATCH];
  if (threadIdx.x < BATCH) { s_sum[threadIdx.x] = 0.f; s_ssq[threadIdx.x] = 0.f; }
  __syncthreads();

  // A operand: bf16 state row (batch = m), 4 K-chunks of 32
  const unsigned short* srow =
      statesbf + (size_t)t * (BATCH * D_STATE) + m * D_STATE;
  us16 av[4];
#pragma unroll
  for (int kc = 0; kc < 4; ++kc) {
    const int ka = kc * 32 + 16 * half;
    us8 lo = *(const us8*)(srow + ka);
    us8 hi = *(const us8*)(srow + ka + 8);
#pragma unroll
    for (int e = 0; e < 8; ++e) { av[kc][e] = lo[e]; av[kc][e + 8] = hi[e]; }
  }

  v8f acc[8];
#pragma unroll
  for (int sub = 0; sub < 8; ++sub) {
    const int nt = wave * 8 + sub;
    v8f a = {0.f, 0.f, 0.f, 0.f, 0.f, 0.f, 0.f, 0.f};
#pragma unroll
    for (int kc = 0; kc < 4; ++kc) {
      us16 bv = *(const us16*)(Cop + ((size_t)(nt * 4 + kc) * 32 + lane) * 16);
      a = wmma_bf16(av[kc], bv, a);
    }
    acc[sub] = a;
  }

  // Epilogue: y = acc + (D+1)*x ; row mean/var via ds_add_f32
  float dcol[8], gcol[8], bcol[8];
#pragma unroll
  for (int sub = 0; sub < 8; ++sub) {
    const int d = wave * 128 + sub * 16 + m;
    dcol[sub] = Dv[d] + 1.0f;
    gcol[sub] = gamma[d];
    bcol[sub] = beta[d];
  }
  float ps[8], pq[8];
#pragma unroll
  for (int r = 0; r < 8; ++r) { ps[r] = 0.f; pq[r] = 0.f; }
#pragma unroll
  for (int sub = 0; sub < 8; ++sub) {
#pragma unroll
    for (int r = 0; r < 8; ++r) {
      const int row = r + 8 * half;                 // batch
      const int d   = wave * 128 + sub * 16 + m;
      float xv = x[((size_t)row * SEQ + t) * D_MODEL + d];
      float y  = acc[sub][r] + dcol[sub] * xv;
      acc[sub][r] = y;
      ps[r] += y;
      pq[r] += y * y;
    }
  }
#pragma unroll
  for (int r = 0; r < 8; ++r) {
    atomicAdd(&s_sum[r + 8 * half], ps[r]);
    atomicAdd(&s_ssq[r + 8 * half], pq[r]);
  }
  __syncthreads();

  const float inv_n = 1.0f / (float)D_MODEL;
#pragma unroll
  for (int r = 0; r < 8; ++r) {
    const int row = r + 8 * half;
    float mu  = s_sum[row] * inv_n;
    float var = s_ssq[row] * inv_n - mu * mu;
    float rs  = rsqrtf(var + LN_EPS);
#pragma unroll
    for (int sub = 0; sub < 8; ++sub) {
      const int d = wave * 128 + sub * 16 + m;
      float o = (acc[sub][r] - mu) * rs * gcol[sub] + bcol[sub];
      out[((size_t)row * SEQ + t) * D_MODEL + d] = o;
    }
  }
}

// ---------------------------------------------------------------------------
extern "C" void kernel_launch(void* const* d_in, const int* in_sizes, int n_in,
                              void* d_out, int out_size, void* d_ws, size_t ws_size,
                              hipStream_t stream) {
  (void)in_sizes; (void)n_in; (void)out_size; (void)ws_size;
  const float* x     = (const float*)d_in[0];
  const float* Am    = (const float*)d_in[1];
  const float* Bm    = (const float*)d_in[2];
  const float* Cm    = (const float*)d_in[3];
  const float* Dv    = (const float*)d_in[4];
  const float* gamma = (const float*)d_in[5];
  const float* beta  = (const float*)d_in[6];
  float* out = (float*)d_out;

  // workspace: xBs f32 (16 MB) | states bf16 (8 MB) | Bop (256 KB) | Cop (256 KB)
  float*          xBs      = (float*)d_ws;
  unsigned short* statesbf = (unsigned short*)(xBs + (size_t)SEQ * BATCH * D_STATE);
  unsigned short* Bop      = statesbf + (size_t)SEQ * BATCH * D_STATE;
  unsigned short* Cop      = Bop + BOP_N;

  kP_pack  <<<256,     256, 0, stream>>>(Bm, Cm, Bop, Cop);
  kA_xproj <<<SEQ / 8, 256, 0, stream>>>(x, Bop, xBs);
  kB_scan  <<<1,       256, 0, stream>>>(Am, xBs, statesbf);
  kC_out_ln<<<SEQ,     256, 0, stream>>>(statesbf, x, Cop, Dv, gamma, beta, out);
}
